// BiWKV6Encoder_80436147520036
// MI455X (gfx1250) — compile-verified, matrix operand order checked by reading, not run on hardware
//
#include <hip/hip_runtime.h>

// ---------------- model constants ----------------
#define BB    4
#define TT    1024
#define CC    512
#define HEADD 64
#define NHEADD 8
#define NLAYERSS 2
#define FFNN  1792
#define TMM   32
#define TDD   64
#define MM    (BB*TT)            // 4096 rows
#define TOTAL ((size_t)BB*TT*CC) // 2097152

typedef __attribute__((ext_vector_type(16))) __bf16 v16bf;
typedef __attribute__((ext_vector_type(8)))  float  v8f;
typedef unsigned int u32x4 __attribute__((ext_vector_type(4)));
typedef unsigned short ushort_t;

union Frag { u32x4 u[2]; v16bf v; };

__device__ __forceinline__ unsigned short f2bf(float f) {
    unsigned int u = __float_as_uint(f);
    u = (u + 0x7FFFu + ((u >> 16) & 1u)) >> 16;
    return (unsigned short)u;
}

// ---------------- one-time weight convert: W[K][N] f32 -> Wt[Npad][K] bf16 ----------------
__global__ void convT_kernel(const float* __restrict__ W, ushort_t* __restrict__ out,
                             int K, int N, int Npad)
{
    int idx = blockIdx.x * 256 + threadIdx.x;
    if (idx >= K * Npad) return;
    int n = idx / K, k = idx - n * K;
    float v = (n < N) ? W[(size_t)k * N + n] : 0.0f;
    out[idx] = f2bf(v);
}

// ---------------- bf16 WMMA GEMM, 64x64 block tile, double-buffered async staging ----------------
// out[M,N] = epi( A[M,K](lda, bf16) @ Bt[Npad][K]^T (bf16) )
// epi: v = act(acc + bias[n]); if(mul) v*=mul; if(resid) v+=resid; store f32 or bf16
// act: 0=none 1=tanh 2=silu 3=sigmoid 4=relu^2 5=exp(-exp(x))
__global__ __launch_bounds__(256) void gemm_bf16_kernel(
    const ushort_t* __restrict__ A, int lda,
    const ushort_t* __restrict__ Bt,
    void* __restrict__ outp,
    const float* __restrict__ bias,
    const float* __restrict__ resid,
    const float* __restrict__ mul,
    int N, int K, int act, int out_bf16)
{
    __shared__ ushort_t lA[2][64 * 32];
    __shared__ ushort_t lB[2][64 * 32];

    const int tid  = threadIdx.x;
    const int lane = tid & 31;
    const int wid  = tid >> 5;
    const int wm   = wid & 3;      // M sub-tile 0..3
    const int wn   = wid >> 2;     // N pair 0..1 (covers n = wn*32 .. wn*32+31)
    const int mblk = blockIdx.y * 64;
    const int nblk = blockIdx.x * 64;

    // staging map: 4 threads per row, 8 bf16 (16B) per thread
    const int srow = tid >> 2;
    const int scol = (tid & 3) * 8;
    const ushort_t* Ag = A  + (size_t)(mblk + srow) * lda + scol;
    const ushort_t* Bg = Bt + (size_t)(nblk + srow) * K   + scol;
    const unsigned ldsA0 = (unsigned)(size_t)&lA[0][srow * 32 + scol];
    const unsigned ldsA1 = (unsigned)(size_t)&lA[1][srow * 32 + scol];
    const unsigned ldsB0 = (unsigned)(size_t)&lB[0][srow * 32 + scol];
    const unsigned ldsB1 = (unsigned)(size_t)&lB[1][srow * 32 + scol];

    v8f acc0 = {}, acc1 = {};
    const int nk = K >> 5;

    // prologue: async-copy tile 0 into buffer 0
    asm volatile("global_load_async_to_lds_b128 %0, %1, off"
                 :: "v"(ldsA0), "v"(Ag) : "memory");
    asm volatile("global_load_async_to_lds_b128 %0, %1, off"
                 :: "v"(ldsB0), "v"(Bg) : "memory");

    for (int kt = 0; kt < nk; ++kt) {
        const int cur  = kt & 1;
        const bool has_next = (kt + 1 < nk);
        if (has_next) {
            // prefetch tile kt+1 into the other buffer while tile kt computes
            const unsigned nA = cur ? ldsA0 : ldsA1;
            const unsigned nB = cur ? ldsB0 : ldsB1;
            const int kn = (kt + 1) << 5;
            asm volatile("global_load_async_to_lds_b128 %0, %1, off"
                         :: "v"(nA), "v"(Ag + kn) : "memory");
            asm volatile("global_load_async_to_lds_b128 %0, %1, off"
                         :: "v"(nB), "v"(Bg + kn) : "memory");
            // async copies complete in order: <=2 outstanding => tile kt resident
            asm volatile("s_wait_asynccnt 2" ::: "memory");
        } else {
            asm volatile("s_wait_asynccnt 0" ::: "memory");
        }
        __syncthreads();

        const ushort_t* bufA = lA[cur];
        const ushort_t* bufB = lB[cur];
        Frag fa, fb0, fb1;
        const int r16 = lane & 15;
        const int kh  = (lane >> 4) << 3;   // K-half select per lane group
        const int ab  = (wm * 16 + r16) * 32 + kh;
        fa.u[0]  = *(const u32x4*)&bufA[ab];
        fa.u[1]  = *(const u32x4*)&bufA[ab + 16];
        const int bb0 = (wn * 32 + r16) * 32 + kh;
        fb0.u[0] = *(const u32x4*)&bufB[bb0];
        fb0.u[1] = *(const u32x4*)&bufB[bb0 + 16];
        const int bb1 = (wn * 32 + 16 + r16) * 32 + kh;
        fb1.u[0] = *(const u32x4*)&bufB[bb1];
        fb1.u[1] = *(const u32x4*)&bufB[bb1 + 16];

        acc0 = __builtin_amdgcn_wmma_f32_16x16x32_bf16(
                   false, fa.v, false, fb0.v, (short)0, acc0, false, false);
        acc1 = __builtin_amdgcn_wmma_f32_16x16x32_bf16(
                   false, fa.v, false, fb1.v, (short)0, acc1, false, false);
        __syncthreads();   // buffer 'cur' free for reuse after this
    }

    // epilogue (both 16x16 accumulators)
    #pragma unroll
    for (int half = 0; half < 2; ++half) {
        v8f a = half ? acc1 : acc0;
        const int colg = nblk + wn * 32 + half * 16 + (lane & 15);
        if (colg < N) {
            #pragma unroll
            for (int r = 0; r < 8; ++r) {
                int rowg = mblk + wm * 16 + r + ((lane >> 4) << 3);
                float v = a[r];
                if (bias) v += bias[colg];
                switch (act) {
                    case 1: v = tanhf(v); break;
                    case 2: v = v / (1.0f + __expf(-v)); break;
                    case 3: v = 1.0f / (1.0f + __expf(-v)); break;
                    case 4: { float t = fmaxf(v, 0.0f); v = t * t; } break;
                    case 5: v = __expf(-__expf(v)); break;
                    default: break;
                }
                size_t oi = (size_t)rowg * N + colg;
                if (mul)   v *= mul[oi];
                if (resid) v += resid[oi];
                if (out_bf16) ((ushort_t*)outp)[oi] = f2bf(v);
                else          ((float*)outp)[oi]    = v;
            }
        }
    }
}

// ---------------- LayerNorm (one row of C=512 per block) ----------------
__global__ __launch_bounds__(256) void ln_kernel(
    const float* __restrict__ in, const float* __restrict__ w,
    const float* __restrict__ b, float* __restrict__ out)
{
    __shared__ float s1[8], s2[8];
    const int row = blockIdx.x;
    const float* x = in + (size_t)row * CC;
    float v0 = x[threadIdx.x], v1 = x[threadIdx.x + 256];
    float s = v0 + v1, sq = v0 * v0 + v1 * v1;
    #pragma unroll
    for (int o = 16; o; o >>= 1) { s += __shfl_xor(s, o, 32); sq += __shfl_xor(sq, o, 32); }
    if ((threadIdx.x & 31) == 0) { s1[threadIdx.x >> 5] = s; s2[threadIdx.x >> 5] = sq; }
    __syncthreads();
    float ts = 0.f, tq = 0.f;
    #pragma unroll
    for (int j = 0; j < 8; ++j) { ts += s1[j]; tq += s2[j]; }
    float mu  = ts / (float)CC;
    float var = tq / (float)CC - mu * mu;
    float inv = rsqrtf(var + 1e-5f);
    out[(size_t)row * CC + threadIdx.x]       = (v0 - mu) * inv * w[threadIdx.x]       + b[threadIdx.x];
    out[(size_t)row * CC + threadIdx.x + 256] = (v1 - mu) * inv * w[threadIdx.x + 256] + b[threadIdx.x + 256];
}

// ---------------- GroupNorm * g -> bf16 ----------------
__global__ __launch_bounds__(256) void gn_kernel(
    const float* __restrict__ y, const float* __restrict__ w,
    const float* __restrict__ b, const float* __restrict__ g,
    ushort_t* __restrict__ out)
{
    const int row  = blockIdx.x;
    const int wid  = threadIdx.x >> 5;   // head (8 waves -> 8 heads)
    const int lane = threadIdx.x & 31;
    size_t base = (size_t)row * CC + wid * HEADD;
    float v0 = y[base + lane], v1 = y[base + lane + 32];
    float s = v0 + v1, sq = v0 * v0 + v1 * v1;
    #pragma unroll
    for (int o = 16; o; o >>= 1) { s += __shfl_xor(s, o, 32); sq += __shfl_xor(sq, o, 32); }
    float mu  = s * (1.0f / 64.0f);
    float var = sq * (1.0f / 64.0f) - mu * mu;
    float inv = rsqrtf(var + 64e-5f);
    int c0 = wid * HEADD + lane, c1 = c0 + 32;
    out[base + lane]      = f2bf(((v0 - mu) * inv * w[c0] + b[c0]) * g[(size_t)row * CC + c0]);
    out[base + lane + 32] = f2bf(((v1 - mu) * inv * w[c1] + b[c1]) * g[(size_t)row * CC + c1]);
}

// ---------------- token-shift mix -> bf16 ----------------
__global__ void mix_kernel(const float* __restrict__ xt, const float* __restrict__ maa,
                           const float* __restrict__ dl, ushort_t* __restrict__ out)
{
    size_t idx = (size_t)blockIdx.x * blockDim.x + threadIdx.x;
    if (idx >= TOTAL) return;
    int c = (int)(idx & (CC - 1));
    int t = (int)((idx >> 9) & (TT - 1));
    float cur  = xt[idx];
    float prev = (t > 0) ? xt[idx - CC] : 0.0f;
    float m = maa[c] + (dl ? dl[idx] : 0.0f);
    out[idx] = f2bf(cur + (prev - cur) * m);
}

// ---------------- add sinusoidal positional encoding ----------------
__global__ void addpe_kernel(const float* __restrict__ x, float* __restrict__ out)
{
    size_t idx = (size_t)blockIdx.x * blockDim.x + threadIdx.x;
    if (idx >= TOTAL) return;
    int c = (int)(idx & (CC - 1));
    int t = (int)((idx >> 9) & (TT - 1));
    int cpair = c & ~1;
    float div = __expf((float)cpair * (-9.210340371976184f / (float)CC)); // -ln(10000)/C
    float ang = (float)t * div;
    float pe  = (c & 1) ? __cosf(ang) : __sinf(ang);
    out[idx] = x[idx] + pe;
}

// ---------------- reverse along T ----------------
__global__ void rev_kernel(const float* __restrict__ in, float* __restrict__ out)
{
    size_t idx = (size_t)blockIdx.x * blockDim.x + threadIdx.x;
    if (idx >= TOTAL) return;
    int c = (int)(idx & (CC - 1));
    int t = (int)((idx >> 9) & (TT - 1));
    size_t b = idx >> 19;
    out[idx] = in[(b << 19) + ((size_t)(TT - 1 - t) << 9) + c];
}

// ---------------- depthwise conv (K=3, SAME) minus identity -> bf16 ----------------
__global__ void dwconv_kernel(const float* __restrict__ x, const float* __restrict__ cw,
                              const float* __restrict__ cb, ushort_t* __restrict__ out)
{
    size_t idx = (size_t)blockIdx.x * blockDim.x + threadIdx.x;
    if (idx >= TOTAL) return;
    int c = (int)(idx & (CC - 1));
    int t = (int)((idx >> 9) & (TT - 1));
    float a = (t > 0)      ? x[idx - CC] : 0.0f;
    float m = x[idx];
    float z = (t < TT - 1) ? x[idx + CC] : 0.0f;
    out[idx] = f2bf(a * cw[c * 3 + 0] + m * cw[c * 3 + 1] + z * cw[c * 3 + 2] + cb[c] - m);
}

// ---------------- gated fwd/bwd combine + length mask ----------------
__global__ void combine_kernel(const float* __restrict__ G, const float* __restrict__ pf,
                               const float* __restrict__ pb, const int* __restrict__ len,
                               float* __restrict__ out)
{
    size_t idx = (size_t)blockIdx.x * blockDim.x + threadIdx.x;
    if (idx >= TOTAL) return;
    int t = (int)((idx >> 9) & (TT - 1));
    int b = (int)(idx >> 19);
    float g = G[idx];
    float v = g * pf[idx] + (1.0f - g) * pb[idx];
    out[idx] = (t < len[b]) ? v : 0.0f;
}

// ---------------- WKV6 sequential scan: one block per (b,head) ----------------
__global__ __launch_bounds__(64) void wkv6_kernel(
    const float* __restrict__ r, const float* __restrict__ k,
    const float* __restrict__ v, const float* __restrict__ w,
    const float* __restrict__ u, float* __restrict__ y)
{
    __shared__ float sr[64], sk[64], sw[64], su[64];
    const int b = blockIdx.x >> 3;
    const int h = blockIdx.x & 7;
    const int j = threadIdx.x;
    su[j] = u[h * 64 + j];
    float S[64];
    #pragma unroll
    for (int i = 0; i < 64; ++i) S[i] = 0.0f;
    for (int t = 0; t < TT; ++t) {
        size_t base = ((((size_t)b * TT + t) * NHEADD) + h) * 64;
        __syncthreads();
        sr[j] = r[base + j]; sk[j] = k[base + j]; sw[j] = w[base + j];
        __syncthreads();
        float vj  = v[base + j];
        float acc = 0.0f;
        #pragma unroll 8
        for (int i = 0; i < 64; ++i) {
            float kv = sk[i] * vj;
            acc = fmaf(sr[i], S[i] + su[i] * kv, acc);
            S[i] = fmaf(sw[i], S[i], kv);
        }
        y[base + j] = acc;
    }
}

// =====================================================================
extern "C" void kernel_launch(void* const* d_in, const int* in_sizes, int n_in,
                              void* d_out, int out_size, void* d_ws, size_t ws_size,
                              hipStream_t stream)
{
    (void)in_sizes; (void)n_in; (void)out_size; (void)ws_size;
    // ---- inputs (setup_inputs dict order) ----
    const float* x_in    = (const float*)d_in[0];
    const int*   lengths = (const int*)  d_in[1];
    const float* ln0_w = (const float*)d_in[2],  *ln0_b = (const float*)d_in[3];
    const float* ln1_w = (const float*)d_in[4],  *ln1_b = (const float*)d_in[5];
    const float* ln2_w = (const float*)d_in[6],  *ln2_b = (const float*)d_in[7];
    const float* maa_x = (const float*)d_in[8];
    const float* maa_w = (const float*)d_in[9],  *maa_k = (const float*)d_in[10];
    const float* maa_v = (const float*)d_in[11], *maa_r = (const float*)d_in[12];
    const float* maa_g = (const float*)d_in[13];
    const float* tm_w1 = (const float*)d_in[14], *tm_w2 = (const float*)d_in[15];
    const float* td_w1 = (const float*)d_in[16], *td_w2 = (const float*)d_in[17];
    const float* time_decay = (const float*)d_in[18];
    const float* time_faaaa = (const float*)d_in[19];
    const float* Wr = (const float*)d_in[20], *Wk = (const float*)d_in[21];
    const float* Wv = (const float*)d_in[22], *Wg = (const float*)d_in[23];
    const float* Wo = (const float*)d_in[24];
    const float* lnx_w = (const float*)d_in[25], *lnx_b = (const float*)d_in[26];
    const float* cm_maa_k = (const float*)d_in[27], *cm_maa_r = (const float*)d_in[28];
    const float* cm_Wk = (const float*)d_in[29], *cm_Wv = (const float*)d_in[30];
    const float* cm_Wr = (const float*)d_in[31];
    const float* gate_w = (const float*)d_in[32], *gate_b = (const float*)d_in[33];
    const float* conv_w = (const float*)d_in[34], *conv_b = (const float*)d_in[35];

    // ---- workspace layout ----
    const size_t N1 = TOTAL;
    float* p = (float*)d_ws;
    float* xcur = p;  p += N1;
    float* pf   = p;  p += N1;
    float* pb   = p;  p += N1;
    float* xt   = p;  p += N1;
    float* dlb  = p;  p += N1;
    float* rb   = p;  p += N1;
    float* kb   = p;  p += N1;
    float* vb   = p;  p += N1;
    float* wdec = p;  p += N1;
    float* gb   = p;  p += N1;
    float* yb   = p;  p += N1;
    float* pbrev= p;  p += N1;
    float* Gbuf = p;  p += N1;
    float* sigb = p;  p += N1;
    ushort_t* q = (ushort_t*)p;
    ushort_t* mixbf = q;  q += N1;
    ushort_t* gnybf = q;  q += N1;
    ushort_t* xresbf= q;  q += N1;
    ushort_t* t5bf  = q;  q += (size_t)MM * 160;
    ushort_t* tdtbf = q;  q += (size_t)MM * TDD;
    ushort_t* ffnbf = q;  q += (size_t)MM * FFNN;
    ushort_t* wp    = q;  // bump allocator for converted weights

    const dim3 eg((unsigned)((TOTAL + 255) / 256)), eb(256);

    // one-time weight convert+transpose+pad
    auto convT = [&](const float* W, int K, int N) -> ushort_t* {
        int Npad = (N + 63) & ~63;
        ushort_t* dst = wp;
        wp += (size_t)Npad * K;
        int total = Npad * K;
        convT_kernel<<<dim3((unsigned)((total + 255) / 256)), 256, 0, stream>>>(W, dst, K, N, Npad);
        return dst;
    };

    struct DLW { ushort_t *tm1, *tm2[5], *td1, *td2, *wk, *wv, *wr, *wg, *wo, *cmk, *cmv, *cmr; };
    DLW wts[4];
    ushort_t* gateT[NLAYERSS];
    for (int dl = 0; dl < 4; ++dl) {
        DLW& W2 = wts[dl];
        W2.tm1 = convT(tm_w1 + (size_t)dl * CC * 160, CC, 160);
        for (int f = 0; f < 5; ++f)
            W2.tm2[f] = convT(tm_w2 + ((size_t)dl * 5 + f) * TMM * CC, TMM, CC);
        W2.td1 = convT(td_w1 + (size_t)dl * CC * TDD, CC, TDD);
        W2.td2 = convT(td_w2 + (size_t)dl * TDD * CC, TDD, CC);
        W2.wk  = convT(Wk + (size_t)dl * CC * CC, CC, CC);
        W2.wv  = convT(Wv + (size_t)dl * CC * CC, CC, CC);
        W2.wr  = convT(Wr + (size_t)dl * CC * CC, CC, CC);
        W2.wg  = convT(Wg + (size_t)dl * CC * CC, CC, CC);
        W2.wo  = convT(Wo + (size_t)dl * CC * CC, CC, CC);
        W2.cmk = convT(cm_Wk + (size_t)dl * CC * FFNN, CC, FFNN);
        W2.cmv = convT(cm_Wv + (size_t)dl * FFNN * CC, FFNN, CC);
        W2.cmr = convT(cm_Wr + (size_t)dl * CC * CC, CC, CC);
    }
    for (int i = 0; i < NLAYERSS; ++i)
        gateT[i] = convT(gate_w + (size_t)i * CC * CC, CC, CC);

    auto gemm = [&](const ushort_t* A, int lda, const ushort_t* Bt, void* out_,
                    const float* bias, const float* resid, const float* mul,
                    int N, int K, int act, int obf) {
        int Npad = (N + 63) & ~63;
        dim3 g((unsigned)(Npad / 64), (unsigned)(MM / 64));
        gemm_bf16_kernel<<<g, 256, 0, stream>>>(A, lda, Bt, out_, bias, resid, mul,
                                                N, K, act, obf);
    };

    // one RWKV6 block operating in-place on xb (f32)
    auto block = [&](float* xb, int d, int i, bool first) {
        const int dl = d * NLAYERSS + i;
        const DLW& W2 = wts[dl];
        const size_t oC = (size_t)dl * CC;
        if (first) ln_kernel<<<MM, 256, 0, stream>>>(xb, ln0_w, ln0_b, xb);
        // ---- time mix ----
        ln_kernel<<<MM, 256, 0, stream>>>(xb, ln1_w + oC, ln1_b + oC, xt);
        mix_kernel<<<eg, eb, 0, stream>>>(xt, maa_x + oC, nullptr, mixbf);
        gemm(mixbf, CC, W2.tm1, t5bf, nullptr, nullptr, nullptr, 160, CC, 1 /*tanh*/, 1);
        const float* maas[5] = { maa_w + oC, maa_k + oC, maa_v + oC, maa_r + oC, maa_g + oC };
        for (int f = 0; f < 5; ++f) {
            gemm(t5bf + f * TMM, 160, W2.tm2[f], dlb, nullptr, nullptr, nullptr, CC, TMM, 0, 0);
            mix_kernel<<<eg, eb, 0, stream>>>(xt, maas[f], dlb, mixbf);
            switch (f) {
                case 0:
                    gemm(mixbf, CC, W2.td1, tdtbf, nullptr, nullptr, nullptr, TDD, CC, 1, 1);
                    gemm(tdtbf, TDD, W2.td2, wdec, time_decay + oC, nullptr, nullptr,
                         CC, TDD, 5 /*exp(-exp)*/, 0);
                    break;
                case 1: gemm(mixbf, CC, W2.wk, kb, nullptr, nullptr, nullptr, CC, CC, 0, 0); break;
                case 2: gemm(mixbf, CC, W2.wv, vb, nullptr, nullptr, nullptr, CC, CC, 0, 0); break;
                case 3: gemm(mixbf, CC, W2.wr, rb, nullptr, nullptr, nullptr, CC, CC, 0, 0); break;
                case 4: gemm(mixbf, CC, W2.wg, gb, nullptr, nullptr, nullptr, CC, CC, 2 /*silu*/, 0); break;
            }
        }
        wkv6_kernel<<<BB * NHEADD, 64, 0, stream>>>(rb, kb, vb, wdec,
                                                    time_faaaa + (size_t)dl * NHEADD * HEADD, yb);
        gn_kernel<<<MM, 256, 0, stream>>>(yb, lnx_w + oC, lnx_b + oC, gb, gnybf);
        gemm(gnybf, CC, W2.wo, xb, nullptr, xb /*residual*/, nullptr, CC, CC, 0, 0);
        // ---- channel mix ----
        ln_kernel<<<MM, 256, 0, stream>>>(xb, ln2_w + oC, ln2_b + oC, xt);
        mix_kernel<<<eg, eb, 0, stream>>>(xt, cm_maa_k + oC, nullptr, mixbf);
        gemm(mixbf, CC, W2.cmk, ffnbf, nullptr, nullptr, nullptr, FFNN, CC, 4 /*relu^2*/, 1);
        mix_kernel<<<eg, eb, 0, stream>>>(xt, cm_maa_r + oC, nullptr, mixbf);
        gemm(mixbf, CC, W2.cmr, sigb, nullptr, nullptr, nullptr, CC, CC, 3 /*sigmoid*/, 0);
        gemm(ffnbf, FFNN, W2.cmv, xb, nullptr, xb /*resid*/, sigb /*mul*/, CC, FFNN, 0, 0);
    };

    // x = x + positional encoding
    addpe_kernel<<<eg, eb, 0, stream>>>(x_in, xcur);

    for (int i = 0; i < NLAYERSS; ++i) {
        // forward branch
        hipMemcpyAsync(pf, xcur, N1 * sizeof(float), hipMemcpyDeviceToDevice, stream);
        block(pf, 0, i, i == 0);
        // backward branch (time reversed)
        rev_kernel<<<eg, eb, 0, stream>>>(xcur, pb);
        block(pb, 1, i, false);
        rev_kernel<<<eg, eb, 0, stream>>>(pb, pbrev);
        // direction gate
        dwconv_kernel<<<eg, eb, 0, stream>>>(xcur, conv_w + (size_t)i * CC * 3,
                                             conv_b + (size_t)i * CC, xresbf);
        gemm(xresbf, CC, gateT[i], Gbuf, gate_b + (size_t)i * CC, nullptr, nullptr,
             CC, CC, 3 /*sigmoid*/, 0);
        float* dst = (i == NLAYERSS - 1) ? (float*)d_out : xcur;
        combine_kernel<<<eg, eb, 0, stream>>>(Gbuf, pf, pbrev, lengths, dst);
    }
}
